// ConvGRU_42803644072243
// MI455X (gfx1250) — matrix operand order repaired and checked by reading, not compiled, not run on hardware
//
#include <hip/hip_runtime.h>

// ---------------- problem constants ----------------
#define B_    8
#define HID_  128
#define INP_  320
#define C_    448      // HID + INP
#define H_    64
#define W_    128
#define P_    64       // spatial positions per block tile (half row)
#define KC_   32       // K per v_wmma_f32_16x16x32_f16
#define NKC_  (C_ / KC_)   // 14 K-chunks
#define AS_   456      // activation k-stride in f16 (448 + 8 pad, keeps 16B align)

typedef __attribute__((ext_vector_type(16))) _Float16 v16h;
typedef __attribute__((ext_vector_type(8)))  float    v8f;

union Frag16 { v16h v; uint4 q[2]; _Float16 e[16]; };

// =====================================================================
// Kernel 1: per-tensor symmetric int8 fake-quant prep.
//   blocks 0..2 : depthwise weights -> fake-quant f32 (q*s)
//   blocks 3..5 : pointwise weights -> integer-valued f16 (q) + scale s
// =====================================================================
__global__ __launch_bounds__(256) void convgru_quant_prep(
    const float* wdz, const float* wdr, const float* wdq,
    const float* wpz, const float* wpr, const float* wpq,
    float* fqwd, _Float16* qwp, float* scales)
{
    __shared__ float red[256];
    __shared__ float s_scale;
    const int t = threadIdx.x;
    const int which = blockIdx.x;  // 0..5

    const float* src; int n;
    if      (which == 0) { src = wdz; n = C_ * 9;    }
    else if (which == 1) { src = wdr; n = C_ * 9;    }
    else if (which == 2) { src = wdq; n = C_ * 9;    }
    else if (which == 3) { src = wpz; n = HID_ * C_; }
    else if (which == 4) { src = wpr; n = HID_ * C_; }
    else                 { src = wpq; n = HID_ * C_; }

    float m = 0.0f;
    for (int i = t; i < n; i += 256) m = fmaxf(m, fabsf(src[i]));
    red[t] = m;
    __syncthreads();
    for (int s = 128; s > 0; s >>= 1) {
        if (t < s) red[t] = fmaxf(red[t], red[t + s]);
        __syncthreads();
    }
    if (t == 0) s_scale = fmaxf(red[0], 1e-8f) / 127.0f;
    __syncthreads();
    const float sc = s_scale;

    if (which < 3) {
        float* dst = fqwd + which * (C_ * 9);
        for (int i = t; i < n; i += 256) {
            float q = rintf(src[i] / sc);          // RNE, matches jnp.round
            q = fminf(fmaxf(q, -128.0f), 127.0f);
            dst[i] = q * sc;
        }
    } else {
        _Float16* dst = qwp + (which - 3) * (HID_ * C_);
        for (int i = t; i < n; i += 256) {
            float q = rintf(src[i] / sc);
            q = fminf(fmaxf(q, -128.0f), 127.0f);
            dst[i] = (_Float16)q;                  // integer, exact in f16
        }
        if (t == 0) scales[which - 3] = sc;
    }
}

// =====================================================================
// Depthwise 3x3 SAME conv of concat(src0[0:128], src1[0:320]) for one
// (batch b, row y, cols x0..x0+63) tile -> f16 activation matrix in LDS,
// POSITION-MAJOR: A[p][k], k-stride AS_ (16B-aligned fragment rows).
// =====================================================================
__device__ __attribute__((always_inline)) inline void dw_fill(
    _Float16* __restrict__ Alds,
    const float* __restrict__ src0,   // channels [0,128)
    const float* __restrict__ src1,   // channels [128,448)
    const float* __restrict__ wd,     // fake-quant dw weights [448][9]
    const float* __restrict__ bd,     // dw bias [448]
    int b, int y, int x0)
{
    const int t  = threadIdx.x;
    const int p  = t & 63;        // position in tile
    const int c0 = t >> 6;        // 0..3
    const int xc = x0 + p;
    _Float16* arow = Alds + p * AS_;

    for (int c = c0; c < C_; c += 4) {
        const float* plane = (c < HID_)
            ? src0 + ((size_t)b * HID_ + c)          * (size_t)(H_ * W_)
            : src1 + ((size_t)b * INP_ + (c - HID_)) * (size_t)(H_ * W_);
        const float* w = wd + c * 9;
        float acc = bd[c];
#pragma unroll
        for (int dy = -1; dy <= 1; ++dy) {
            const int yy = y + dy;
            if ((unsigned)yy < (unsigned)H_) {
                const float* row = plane + yy * W_;
#pragma unroll
                for (int dx = -1; dx <= 1; ++dx) {
                    const int xx = xc + dx;
                    if ((unsigned)xx < (unsigned)W_)
                        acc += w[(dy + 1) * 3 + (dx + 1)] * row[xx];
                }
            }
        }
        arow[c] = (_Float16)acc;
    }
}

// =====================================================================
// GEMM core: acc[4] += Wq[128x448] x A[448x64] via v_wmma_f32_16x16x32_f16.
// 8 waves: wave w owns M-tile w (rows 16w..16w+15), all 4 N-tiles.
// Weight fragments for all 14 K-chunks preloaded into VGPRs from global
// (L2-resident, 2 x b128 per chunk per lane). K-loop: pure ds_load_b128
// + wmma, no barriers.
// =====================================================================
__device__ __attribute__((always_inline)) inline void gemm_accum(
    const _Float16* __restrict__ Alds,   // [64][AS_] f16 activations
    const Frag16 afr[NKC_],              // preloaded weight fragments
    v8f acc[4])
{
    const int t    = threadIdx.x;
    const int lane = t & 31;
    const int l16  = lane & 15;
    const int hi   = lane >> 4;          // 0 or 1 (lane half)

#pragma unroll
    for (int kc = 0; kc < NKC_; ++kc) {
#pragma unroll
        for (int nt = 0; nt < 4; ++nt) {
            Frag16 bb;
            const int n = nt * 16 + l16;
            const uint4* src = (const uint4*)(Alds + n * AS_ + kc * KC_ + hi * 16);
            bb.q[0] = src[0];            // 16 contiguous K values: 2x ds_load_b128
            bb.q[1] = src[1];
            acc[nt] = __builtin_amdgcn_wmma_f32_16x16x32_f16(
                false, afr[kc].v, false, bb.v, (short)0, acc[nt], false, false);
        }
    }
}

__device__ __attribute__((always_inline)) inline void preload_wfrags(
    const _Float16* __restrict__ qw,  // [128][448] f16 integer weights
    Frag16 afr[NKC_])
{
    const int t    = threadIdx.x;
    const int lane = t & 31;
    const int l16  = lane & 15;
    const int hi   = lane >> 4;
    const int m    = (t >> 5) * 16 + l16;     // weight row this lane carries
    const _Float16* wrow = qw + m * C_;
#pragma unroll
    for (int kc = 0; kc < NKC_; ++kc) {
        // A 16x32 layout (matches B): lane half 0 -> K {0-15}, half 1 -> {16-31}
        afr[kc].q[0] = *(const uint4*)(wrow + kc * KC_ + hi * 16);
        afr[kc].q[1] = *(const uint4*)(wrow + kc * KC_ + hi * 16 + 8);
    }
}

// =====================================================================
// Pass 1: z and r gates.  z -> z_ws,  r*h -> rh_ws.
// grid = B*H*2 blocks; block = (b, y, half-row) tile of 64 px.
// =====================================================================
__global__ __launch_bounds__(256) void convgru_pass1(
    const float* __restrict__ h, const float* __restrict__ x,
    const float* __restrict__ fqwd, const _Float16* __restrict__ qwp,
    const float* __restrict__ scales,
    const float* __restrict__ bdz, const float* __restrict__ bpz,
    const float* __restrict__ bdr, const float* __restrict__ bpr,
    float* __restrict__ z_ws, float* __restrict__ rh_ws)
{
    extern __shared__ char smem[];
    _Float16* Alds = (_Float16*)smem;     // 64 * 456 * 2 = 58368 B

    const int blk  = blockIdx.x;
    const int half = blk & 1;
    const int y    = (blk >> 1) & (H_ - 1);
    const int b    = blk >> 7;
    const int x0   = half * P_;

    const int t = threadIdx.x;
    const int wv = t >> 5, lane = t & 31, l16 = lane & 15, hi = lane >> 4;

    for (int g = 0; g < 2; ++g) {
        const float*    wd = fqwd + g * (C_ * 9);
        const _Float16* qw = qwp  + g * (HID_ * C_);
        const float*    bd = (g == 0) ? bdz : bdr;
        const float*    bp = (g == 0) ? bpz : bpr;
        const float     sc = scales[g];

        Frag16 afr[NKC_];
        preload_wfrags(qw, afr);          // overlaps with barrier below

        __syncthreads();                  // previous gate's Alds reads done
        dw_fill(Alds, h, x, wd, bd, b, y, x0);
        __syncthreads();                  // Alds ready

        v8f acc[4] = {};
        gemm_accum(Alds, afr, acc);

        // epilogue: sigmoid; z store, or r*h store
#pragma unroll
        for (int nt = 0; nt < 4; ++nt) {
            const int n = x0 + nt * 16 + l16;
#pragma unroll
            for (int v = 0; v < 8; ++v) {
                const int m = wv * 16 + v + 8 * hi;
                const float val = acc[nt][v] * sc + bp[m];
                const float sg  = 1.0f / (1.0f + __expf(-val));
                const size_t idx = (((size_t)b * HID_ + m) * H_ + y) * W_ + n;
                if (g == 0) z_ws[idx] = sg;
                else        rh_ws[idx] = sg * h[idx];
            }
        }
    }
}

// =====================================================================
// Pass 2: q gate from concat(r*h, x), then out = (1-z)*h + z*q.
// =====================================================================
__global__ __launch_bounds__(256) void convgru_pass2(
    const float* __restrict__ h, const float* __restrict__ x,
    const float* __restrict__ fqwd, const _Float16* __restrict__ qwp,
    const float* __restrict__ scales,
    const float* __restrict__ bdq, const float* __restrict__ bpq,
    const float* __restrict__ z_ws, const float* __restrict__ rh_ws,
    float* __restrict__ out)
{
    extern __shared__ char smem[];
    _Float16* Alds = (_Float16*)smem;

    const int blk  = blockIdx.x;
    const int half = blk & 1;
    const int y    = (blk >> 1) & (H_ - 1);
    const int b    = blk >> 7;
    const int x0   = half * P_;

    const int t = threadIdx.x;
    const int wv = t >> 5, lane = t & 31, l16 = lane & 15, hi = lane >> 4;

    Frag16 afr[NKC_];
    preload_wfrags(qwp + 2 * (HID_ * C_), afr);

    dw_fill(Alds, rh_ws, x, fqwd + 2 * (C_ * 9), bdq, b, y, x0);
    __syncthreads();

    v8f acc[4] = {};
    gemm_accum(Alds, afr, acc);

    const float sc = scales[2];
#pragma unroll
    for (int nt = 0; nt < 4; ++nt) {
        const int n = x0 + nt * 16 + l16;
#pragma unroll
        for (int v = 0; v < 8; ++v) {
            const int m = wv * 16 + v + 8 * hi;
            const float val = acc[nt][v] * sc + bpq[m];
            const float q   = tanhf(val);
            const size_t idx = (((size_t)b * HID_ + m) * H_ + y) * W_ + n;
            const float z = z_ws[idx];
            out[idx] = (1.0f - z) * h[idx] + z * q;
        }
    }
}

// =====================================================================
extern "C" void kernel_launch(void* const* d_in, const int* in_sizes, int n_in,
                              void* d_out, int out_size, void* d_ws, size_t ws_size,
                              hipStream_t stream) {
    const float* h   = (const float*)d_in[0];
    const float* x   = (const float*)d_in[1];
    const float* wdz = (const float*)d_in[2];  const float* bdz = (const float*)d_in[3];
    const float* wpz = (const float*)d_in[4];  const float* bpz = (const float*)d_in[5];
    const float* wdr = (const float*)d_in[6];  const float* bdr = (const float*)d_in[7];
    const float* wpr = (const float*)d_in[8];  const float* bpr = (const float*)d_in[9];
    const float* wdq = (const float*)d_in[10]; const float* bdq = (const float*)d_in[11];
    const float* wpq = (const float*)d_in[12]; const float* bpq = (const float*)d_in[13];

    char* ws = (char*)d_ws;
    const size_t plane_bytes = (size_t)B_ * HID_ * H_ * W_ * sizeof(float); // 33.5 MB
    const size_t zoff  = 0;
    const size_t rhoff = plane_bytes;
    const size_t fqoff = 2 * plane_bytes;
    const size_t qwoff = fqoff + (size_t)3 * C_ * 9 * sizeof(float);   // 16B aligned
    const size_t scoff = qwoff + (size_t)3 * HID_ * C_ * sizeof(_Float16);

    float*    z_ws   = (float*)(ws + zoff);
    float*    rh_ws  = (float*)(ws + rhoff);
    float*    fqwd   = (float*)(ws + fqoff);
    _Float16* qwp    = (_Float16*)(ws + qwoff);
    float*    scales = (float*)(ws + scoff);

    convgru_quant_prep<<<6, 256, 0, stream>>>(wdz, wdr, wdq, wpz, wpr, wpq,
                                              fqwd, qwp, scales);

    const size_t shmem = (size_t)P_ * AS_ * sizeof(_Float16); // 58368 B
    const int nblk = B_ * H_ * 2; // 1024

    convgru_pass1<<<nblk, 256, shmem, stream>>>(h, x, fqwd, qwp, scales,
                                                bdz, bpz, bdr, bpr, z_ws, rh_ws);
    convgru_pass2<<<nblk, 256, shmem, stream>>>(h, x, fqwd, qwp, scales,
                                                bdq, bpq, z_ws, rh_ws,
                                                (float*)d_out);
}
